// LocalFeatue_55473797595431
// MI455X (gfx1250) — matrix operand order repaired and checked by reading, not compiled, not run on hardware
//
#include <hip/hip_runtime.h>
#include <hip/hip_bf16.h>

// ---------------- problem constants ----------------
#define B_    4
#define N_    4096
#define C1_   58
#define K_    32
#define CIN_  64
#define RAD2  0.09f
#define EPS_  1e-5f
#define P_    (B_ * N_ * K_)     // 524288 pixels
#define SPREAD 256               // atomic-contention spreading for stats

typedef _Float16 half_t;
typedef __attribute__((ext_vector_type(16))) _Float16 v16h;
typedef __attribute__((ext_vector_type(8)))  _Float16 v8h;
typedef __attribute__((ext_vector_type(8)))  float    v8f;
typedef __attribute__((ext_vector_type(4)))  int      v4i;

typedef __attribute__((address_space(1))) v4i g_v4i;   // global (device) int4
typedef __attribute__((address_space(3))) v4i s_v4i;   // LDS (shared) int4

// ---------------- 1) ball query: first K ascending-index hits within radius
__global__ __launch_bounds__(256)
void ball_query_kernel(const float* __restrict__ xyz, int* __restrict__ inds) {
    __shared__ float sx[N_], sy[N_], sz[N_];          // 48 KB of the 320 KB WGP LDS
    const int b  = blockIdx.x / (N_ / 256);
    const int n0 = (blockIdx.x % (N_ / 256)) * 256;
    const float* xb = xyz + (size_t)b * N_ * 3;
    for (int i = threadIdx.x; i < N_; i += 256) {
        sx[i] = xb[i * 3 + 0];
        sy[i] = xb[i * 3 + 1];
        sz[i] = xb[i * 3 + 2];
    }
    __syncthreads();
    const int n = n0 + threadIdx.x;
    const float qx = sx[n], qy = sy[n], qz = sz[n];
    int* op = inds + ((size_t)b * N_ + n) * K_;
    int cnt = 0, first = n;   // self always qualifies (d2 == 0)
    for (int j = 0; j < N_; ++j) {
        float dx = sx[j] - qx, dy = sy[j] - qy, dz = sz[j] - qz;
        float d2 = dx * dx + dy * dy + dz * dz;
        if (d2 <= RAD2) {
            if (cnt == 0) first = j;
            op[cnt] = j;
            if (++cnt >= K_) break;   // ascending-index scan: first K is final
        }
    }
    for (int t = cnt; t < K_; ++t) op[t] = first;
}

// ---------------- 2) gather + concat -> fp16 activations [P][64]
__global__ __launch_bounds__(256)
void gather_kernel(const float* __restrict__ xyz, const float* __restrict__ feat,
                   const int* __restrict__ inds, half_t* __restrict__ out) {
    const int p = blockIdx.x * 256 + threadIdx.x;      // < P_
    const int b = p / (N_ * K_);
    const int n = (p / K_) % N_;
    const int ii = inds[p];
    const float* cb = xyz + ((size_t)b * N_ + n) * 3;
    const float* nb = xyz + ((size_t)b * N_ + ii) * 3;
    half_t* o = out + (size_t)p * CIN_;
    const float cx = cb[0], cy = cb[1], cz = cb[2];
    o[0] = (half_t)cx;            o[1] = (half_t)cy;            o[2] = (half_t)cz;
    o[3] = (half_t)(nb[0] - cx);  o[4] = (half_t)(nb[1] - cy);  o[5] = (half_t)(nb[2] - cz);
    const float* f = feat + ((size_t)b * N_ + ii) * C1_;
#pragma unroll
    for (int c = 0; c < C1_; ++c) o[6 + c] = (half_t)f[c];
}

// ---------------- small helpers ----------------
__global__ void w2h_kernel(const float* __restrict__ w, half_t* __restrict__ o, int n) {
    int i = blockIdx.x * 256 + threadIdx.x;
    if (i < n) o[i] = (half_t)w[i];
}
__global__ void zero_stats_kernel(double* __restrict__ s, int n) {
    int i = blockIdx.x * 256 + threadIdx.x;
    if (i < n) s[i] = 0.0;
}

// ---------------- 3) fused GEMM:  out[p][o] = sum_c relu(norm(in[p][c])) * W[o][c]
// One wave -> 16-pixel x COUT tile via v_wmma_f32_16x16x32_f16.
// Weights staged global->LDS through the CDNA5 async-to-LDS path (ASYNCcnt).
// Also accumulates GroupNorm sum / sumsq of THIS layer's (pre-norm) output.
template <int COUT, bool NORM_IN>
__global__ __launch_bounds__(256)
void gemm_kernel(const half_t* __restrict__ in, const half_t* __restrict__ Wh,
                 const float* __restrict__ nsc, const float* __restrict__ nsh,
                 half_t* __restrict__ out, double* __restrict__ statsP) {
    __shared__ __align__(64) half_t sW[COUT * CIN_];          // 8 or 16 KB
    constexpr int TOT = COUT * CIN_;
#if __has_builtin(__builtin_amdgcn_global_load_async_to_lds_b128)
    // GLOBAL_LOAD_ASYNC_TO_LDS_B128: no VGPR round-trip, tracked by ASYNCcnt
    for (int i = threadIdx.x * 8; i < TOT; i += 256 * 8)
        __builtin_amdgcn_global_load_async_to_lds_b128(
            (g_v4i*)(Wh + i), (s_v4i*)&sW[i], 0, 0);
#if __has_builtin(__builtin_amdgcn_s_wait_asynccnt)
    __builtin_amdgcn_s_wait_asynccnt(0);
#else
    asm volatile("s_wait_asynccnt 0x0" ::: "memory");
#endif
#else
    for (int i = threadIdx.x * 8; i < TOT; i += 256 * 8)
        *(v8h*)&sW[i] = *(const v8h*)&Wh[i];
#endif
    __syncthreads();

    const int wave = threadIdx.x >> 5;
    const int lane = threadIdx.x & 31;
    const int r    = lane & 15;       // A-row / B-col within tile
    const int hi   = lane >> 4;
    const int tile = blockIdx.x * 8 + wave;
    const size_t pBase = (size_t)tile * 16;
    const int b = (int)(pBase / (N_ * K_));   // 128-pixel blocks never cross batch

    // ---- A fragments (16 pixels x 64 ch), GroupNorm+ReLU of previous layer fused in
    const half_t* arow = in + (pBase + r) * CIN_;
    v16h afrag[2];
#pragma unroll
    for (int f = 0; f < 2; ++f) {
        const int c0 = f * 32 + hi * 8;          // ISA: A lane holds K {c0..c0+7, c0+16..c0+23}
        v8h lo = *(const v8h*)&arow[c0];
        v8h hh = *(const v8h*)&arow[c0 + 16];
        if (NORM_IN) {
#pragma unroll
            for (int j = 0; j < 8; ++j) {
                float x = (float)lo[j];
                x = fmaxf(x * nsc[b * CIN_ + c0 + j] + nsh[b * CIN_ + c0 + j], 0.f);
                lo[j] = (half_t)x;
                float y = (float)hh[j];
                y = fmaxf(y * nsc[b * CIN_ + c0 + 16 + j] + nsh[b * CIN_ + c0 + 16 + j], 0.f);
                hh[j] = (half_t)y;
            }
        }
#pragma unroll
        for (int j = 0; j < 8; ++j) { afrag[f][j] = lo[j]; afrag[f][8 + j] = hh[j]; }
    }

    constexpr int NG = COUT / 32;                // GroupNorm groups of this layer
    float gs1[NG] = {}, gs2[NG] = {};

#pragma unroll
    for (int co = 0; co < COUT / 16; ++co) {
        v8f acc = {};
#pragma unroll
        for (int f = 0; f < 2; ++f) {
            // B lane holds K {hi*8..hi*8+7, 16+hi*8..16+hi*8+7} of W[o][f*32 + ...]
            const half_t* bp = &sW[(co * 16 + r) * CIN_ + f * 32 + hi * 8];
            v8h blo = *(const v8h*)bp;
            v8h bhi = *(const v8h*)(bp + 16);
            v16h bfrag = __builtin_shufflevector(blo, bhi, 0, 1, 2, 3, 4, 5, 6, 7, 8, 9, 10, 11, 12, 13, 14, 15);
            acc = __builtin_amdgcn_wmma_f32_16x16x32_f16(
                false, afrag[f], false, bfrag, (short)0, acc, false, false);
        }
        const int col = co * 16 + r;
        float s1 = 0.f, s2 = 0.f;
#pragma unroll
        for (int j = 0; j < 8; ++j) {
            const int m = j + 8 * hi;            // D: VGPR j -> row j + 8*hi
            out[(pBase + m) * COUT + col] = (half_t)acc[j];
            s1 += acc[j];
            s2 += acc[j] * acc[j];
        }
        gs1[co >> 1] += s1;                      // 2 co-tiles per 32-channel group
        gs2[co >> 1] += s2;
    }

    // ---- wave-reduce stats, spread-bucketed f64 atomics (GLOBAL_ATOMIC_ADD_F64)
#pragma unroll
    for (int g = 0; g < NG; ++g) {
        float s1 = gs1[g], s2 = gs2[g];
#pragma unroll
        for (int m = 16; m; m >>= 1) {
            s1 += __shfl_xor(s1, m, 32);
            s2 += __shfl_xor(s2, m, 32);
        }
        if (lane == 0) {
            double* d = statsP + ((size_t)(b * 4 + g) * SPREAD + (blockIdx.x & (SPREAD - 1))) * 2;
            atomicAdd(d + 0, (double)s1);
            atomicAdd(d + 1, (double)s2);
        }
    }
}

// ---------------- 4) stats -> per-(b,channel) scale/shift
__global__ void finalize_kernel(const double* __restrict__ statsP,
                                const float* __restrict__ gamma, const float* __restrict__ beta,
                                float* __restrict__ nsc, float* __restrict__ nsh, int cout) {
    const int tid = blockIdx.x * 128 + threadIdx.x;
    if (tid >= B_ * cout) return;
    const int b = tid / cout, c = tid % cout, g = c >> 5;
    const double* base = statsP + ((size_t)(b * 4 + g) * SPREAD) * 2;
    double s1 = 0.0, s2 = 0.0;
    for (int s = 0; s < SPREAD; ++s) { s1 += base[s * 2]; s2 += base[s * 2 + 1]; }
    const double cnt  = 32.0 * K_ * N_;          // 32 ch * K * N elements per group
    const double mean = s1 / cnt;
    const double var  = s2 / cnt - mean * mean;
    const float  scl  = gamma[c] * rsqrtf((float)var + EPS_);
    nsc[tid] = scl;
    nsh[tid] = beta[c] - (float)mean * scl;
}

// ---------------- 5) norm2 + relu + max over K, transposed store (B,128,N)
__global__ __launch_bounds__(128)
void max_kernel(const half_t* __restrict__ x, const float* __restrict__ nsc,
                const float* __restrict__ nsh, float* __restrict__ out) {
    const int bn = blockIdx.x;                    // b*N + n
    const int b = bn / N_, n = bn % N_;
    const int c = threadIdx.x;                    // 128 channels
    const float sc = nsc[b * 128 + c], sh = nsh[b * 128 + c];
    const half_t* base = x + (size_t)bn * K_ * 128 + c;
    float m = 0.f;                                // ReLU floor
#pragma unroll
    for (int k = 0; k < K_; ++k)
        m = fmaxf(m, (float)base[k * 128] * sc + sh);
    out[((size_t)b * 128 + c) * N_ + n] = m;
}

// ---------------- driver ----------------
extern "C" void kernel_launch(void* const* d_in, const int* in_sizes, int n_in,
                              void* d_out, int out_size, void* d_ws, size_t ws_size,
                              hipStream_t stream) {
    const float* feature = (const float*)d_in[0];
    const float* xyz     = (const float*)d_in[1];
    const float* W0 = (const float*)d_in[2];
    const float* g0 = (const float*)d_in[3];
    const float* b0 = (const float*)d_in[4];
    const float* W1 = (const float*)d_in[5];
    const float* g1 = (const float*)d_in[6];
    const float* b1 = (const float*)d_in[7];
    const float* W2 = (const float*)d_in[8];
    const float* g2 = (const float*)d_in[9];
    const float* b2 = (const float*)d_in[10];

    char* ws = (char*)d_ws;
    size_t off = 0;
    auto take = [&](size_t bytes) -> void* {
        void* p = ws + off;
        off = (off + bytes + 255) & ~(size_t)255;
        return p;
    };
    int*    inds  = (int*)   take((size_t)P_ * sizeof(int));        //   2 MB
    half_t* bufA  = (half_t*)take((size_t)P_ * 64  * 2);            //  64 MB (x0, reused for x2)
    half_t* bufB  = (half_t*)take((size_t)P_ * 64  * 2);            //  64 MB (x1)
    half_t* bufC  = (half_t*)take((size_t)P_ * 128 * 2);            // 128 MB (x3)
    half_t* Wh0   = (half_t*)take(64  * 64 * 2);
    half_t* Wh1   = (half_t*)take(64  * 64 * 2);
    half_t* Wh2   = (half_t*)take(128 * 64 * 2);
    double* stats = (double*)take((size_t)3 * B_ * 4 * SPREAD * 2 * sizeof(double));
    double* st0 = stats, *st1 = stats + (size_t)B_ * 4 * SPREAD * 2,
          * st2 = stats + (size_t)2 * B_ * 4 * SPREAD * 2;
    float* nsc0 = (float*)take(B_ * 64  * sizeof(float));
    float* nsh0 = (float*)take(B_ * 64  * sizeof(float));
    float* nsc1 = (float*)take(B_ * 64  * sizeof(float));
    float* nsh1 = (float*)take(B_ * 64  * sizeof(float));
    float* nsc2 = (float*)take(B_ * 128 * sizeof(float));
    float* nsh2 = (float*)take(B_ * 128 * sizeof(float));

    // neighbor search + gather/concat (fp16 activations)
    ball_query_kernel<<<B_ * (N_ / 256), 256, 0, stream>>>(xyz, inds);
    gather_kernel<<<P_ / 256, 256, 0, stream>>>(xyz, feature, inds, bufA);

    // weights -> fp16, zero stats buckets (must happen every launch: graph replay)
    w2h_kernel<<<(64 * 64 + 255) / 256, 256, 0, stream>>>(W0, Wh0, 64 * 64);
    w2h_kernel<<<(64 * 64 + 255) / 256, 256, 0, stream>>>(W1, Wh1, 64 * 64);
    w2h_kernel<<<(128 * 64 + 255) / 256, 256, 0, stream>>>(W2, Wh2, 128 * 64);
    const int nStats = 3 * B_ * 4 * SPREAD * 2;
    zero_stats_kernel<<<(nStats + 255) / 256, 256, 0, stream>>>(stats, nStats);

    const int gemmGrid = P_ / 128;   // 8 waves/block, 16 pixels/wave
    // layer 0: raw input, collect stats0
    gemm_kernel<64, false><<<gemmGrid, 256, 0, stream>>>(bufA, Wh0, nullptr, nullptr, bufB, st0);
    finalize_kernel<<<(B_ * 64 + 127) / 128, 128, 0, stream>>>(st0, g0, b0, nsc0, nsh0, 64);
    // layer 1: norm0+relu fused on load, collect stats1
    gemm_kernel<64, true><<<gemmGrid, 256, 0, stream>>>(bufB, Wh1, nsc0, nsh0, bufA, st1);
    finalize_kernel<<<(B_ * 64 + 127) / 128, 128, 0, stream>>>(st1, g1, b1, nsc1, nsh1, 64);
    // layer 2: norm1+relu fused on load, collect stats2
    gemm_kernel<128, true><<<gemmGrid, 256, 0, stream>>>(bufA, Wh2, nsc1, nsh1, bufC, st2);
    finalize_kernel<<<(B_ * 128 + 127) / 128, 128, 0, stream>>>(st2, g2, b2, nsc2, nsh2, 128);
    // norm2 + relu + max over K -> (B,128,N) fp32
    max_kernel<<<B_ * N_, 128, 0, stream>>>(bufC, nsc2, nsh2, (float*)d_out);
}
